// MaskedOmicsFeatureDecoder_76012331204955
// MI455X (gfx1250) — matrix operand order repaired and checked by reading, not compiled, not run on hardware
//
#include <hip/hip_runtime.h>
#include <hip/hip_bf16.h>

// ---- problem constants (from reference) ----
#define B_ROWS 8192
#define NP     2048
#define NA     256
#define NC     32
#define NO     20000
#define NPA    (NP + NA)          // 2304
#define KDIM   (NP + NA + NC)     // 2336 (= 73 * 32, multiple of 32)

// ---- GEMM tiling ----
#define TM 128
#define TN 256
#define TK 32
#define KSTEPS (KDIM / TK)        // 73

typedef __attribute__((ext_vector_type(16))) __bf16 v16bf;
typedef __attribute__((ext_vector_type(8)))  __bf16 v8bf;
typedef __attribute__((ext_vector_type(8)))  float  v8f;

// Async global->LDS DMA path (gfx1250); guarded so compile can't regress.
// Probe (round 2 diagnostic) showed the builtin takes `int vector_size(16)*`
// pointer parameters.
#if defined(__gfx1250__) && \
    __has_builtin(__builtin_amdgcn_global_load_async_to_lds_b128) && \
    __has_builtin(__builtin_amdgcn_s_wait_asynccnt)
#define USE_ASYNC_LDS 1
typedef int v4i __attribute__((vector_size(4 * sizeof(int))));
#else
#define USE_ASYNC_LDS 0
#endif

// -------------------------------------------------------------------------
// Kernel 1: pack zi = [z | cat_covariates_embed] -> bf16, row-major [B x K]
// -------------------------------------------------------------------------
__global__ void pack_z_kernel(const float* __restrict__ z,
                              const float* __restrict__ cat,
                              __bf16* __restrict__ Zb) {
    size_t i = (size_t)blockIdx.x * blockDim.x + threadIdx.x;
    size_t total = (size_t)B_ROWS * KDIM;
    if (i >= total) return;
    int b = (int)(i / KDIM);
    int k = (int)(i % KDIM);
    float v = (k < NPA) ? z[(size_t)b * NPA + k]
                        : cat[(size_t)b * NC + (k - NPA)];
    Zb[i] = (__bf16)v;
}

// -------------------------------------------------------------------------
// Kernel 2: build Wc = concat(W_mask*mask, W_addon*addon_mask, W_cov),
// convert to bf16 and TRANSPOSE to n-major layout Wt[n][k] (K contiguous per
// column -> matches WMMA 16-bit B-fragment layout). LDS 32x32 tile transpose.
// 20000 = 625*32 exactly; 2336 = 73*32 exactly, so no guards needed.
// -------------------------------------------------------------------------
__global__ __launch_bounds__(256) void pack_wt_kernel(
        const float* __restrict__ mask,
        const float* __restrict__ addon_mask,
        const float* __restrict__ W_mask,
        const float* __restrict__ W_addon,
        const float* __restrict__ W_cov,
        __bf16* __restrict__ Wt) {
    __shared__ float s[32][33];
    const int tx = threadIdx.x;          // 0..31
    const int ty = threadIdx.y;          // 0..7
    const int n0 = blockIdx.x * 32;
    const int k0 = blockIdx.y * 32;

    #pragma unroll
    for (int r = 0; r < 4; ++r) {
        int k = k0 + ty + r * 8;
        int n = n0 + tx;
        float w;
        if (k < NP) {
            size_t idx = (size_t)k * NO + n;
            w = W_mask[idx] * mask[idx];
        } else if (k < NPA) {
            size_t idx = (size_t)(k - NP) * NO + n;
            w = W_addon[idx] * addon_mask[idx];
        } else {
            w = W_cov[(size_t)(k - NPA) * NO + n];
        }
        s[ty + r * 8][tx] = w;           // s[k_local][n_local]
    }
    __syncthreads();
    #pragma unroll
    for (int r = 0; r < 4; ++r) {
        int n = n0 + ty + r * 8;
        int k = k0 + tx;
        Wt[(size_t)n * KDIM + k] = (__bf16)s[tx][ty + r * 8];
    }
}

// -------------------------------------------------------------------------
// Kernel 3: bf16 WMMA GEMM. Block tile 128x256, 8 waves (2x4), each wave a
// 64x64 patch = 4x4 tiles of v_wmma_f32_16x16x32_bf16.
// Double-buffered LDS (48KB) with async global->LDS DMA prefetch: each wave
// issues exactly 6 async b128 loads per tile, so s_wait_asynccnt(6) retires
// the current buffer while the next buffer's DMA overlaps the 16 WMMAs.
// -------------------------------------------------------------------------
__global__ __launch_bounds__(256) void gemm_kernel(
        const __bf16* __restrict__ Zb,   // [B x K] row-major
        const __bf16* __restrict__ Wt,   // [NO x K] n-major
        float* __restrict__ out) {       // [B x NO] logits
    __shared__ __align__(16) __bf16 As[2][TM * TK];   // 2 x 8 KB
    __shared__ __align__(16) __bf16 Bs[2][TN * TK];   // 2 x 16 KB

    const int tid  = threadIdx.x;
    const int lane = tid & 31;
    const int wave = tid >> 5;
    const int wm   = wave & 1;           // 0..1
    const int wn   = wave >> 1;          // 0..3
    const int m0   = blockIdx.y * TM;
    const int n0   = blockIdx.x * TN;
    const int lr   = lane & 15;
    const int hi   = lane >> 4;          // 0 or 1

    v8f acc[4][4] = {};

    // Stage one K-tile (TM x TK of A, TN x TK of B) into LDS buffer `buf`.
    // 2 A-loads + 4 B-loads of 16B per thread = 6 per wave slot.
    auto issue_tile = [&](int buf, int k0) {
        #pragma unroll
        for (int it = 0; it < 2; ++it) {
            int idx = tid + it * 256;            // 0..511
            int row = idx >> 2;
            int ch  = idx & 3;
            const __bf16* g = Zb + (size_t)(m0 + row) * KDIM + k0 + ch * 8;
            __bf16* l = &As[buf][row * TK + ch * 8];
#if USE_ASYNC_LDS
            __builtin_amdgcn_global_load_async_to_lds_b128(
                (v4i*)g, (v4i*)l, 0, 0);
#else
            *(v8bf*)l = *(const v8bf*)g;
#endif
        }
        #pragma unroll
        for (int it = 0; it < 4; ++it) {
            int idx  = tid + it * 256;           // 0..1023
            int nrow = idx >> 2;
            int ch   = idx & 3;
            int n    = n0 + nrow;
            __bf16* l = &Bs[buf][nrow * TK + ch * 8];
#if USE_ASYNC_LDS
            // Lanes with n >= NO skip; their LDS slots stay stale but feed
            // only accumulator columns that are never stored.
            if (n < NO)
                __builtin_amdgcn_global_load_async_to_lds_b128(
                    (v4i*)(Wt + (size_t)n * KDIM + k0 + ch * 8),
                    (v4i*)l, 0, 0);
#else
            v8bf v = {};
            if (n < NO)
                v = *(const v8bf*)(Wt + (size_t)n * KDIM + k0 + ch * 8);
            *(v8bf*)l = v;
#endif
        }
    };

    issue_tile(0, 0);

    for (int kt = 0; kt < KSTEPS; ++kt) {
        const int cur = kt & 1;
        if (kt + 1 < KSTEPS)
            issue_tile(cur ^ 1, (kt + 1) * TK);
#if USE_ASYNC_LDS
        if (kt + 1 < KSTEPS)
            __builtin_amdgcn_s_wait_asynccnt(6);  // retire current, keep next in flight
        else
            __builtin_amdgcn_s_wait_asynccnt(0);
#endif
        __syncthreads();

        // ---- gather per-lane fragments (ISA 16-bit A/B VGPR layouts) ----
        v16bf afrag[4], bfrag[4];
        #pragma unroll
        for (int mt = 0; mt < 4; ++mt) {
            const __bf16* base = &As[cur][(wm * 64 + mt * 16 + lr) * TK];
            v8bf lo = *(const v8bf*)(base + hi * 8);        // K 0..7 | 8..15
            v8bf up = *(const v8bf*)(base + 16 + hi * 8);   // K 16..23 | 24..31
            afrag[mt] = __builtin_shufflevector(lo, up,
                0, 1, 2, 3, 4, 5, 6, 7, 8, 9, 10, 11, 12, 13, 14, 15);
        }
        #pragma unroll
        for (int nt = 0; nt < 4; ++nt) {
            const __bf16* base = &Bs[cur][(wn * 64 + nt * 16 + lr) * TK];
            v8bf lo = *(const v8bf*)(base + hi * 8);        // K 0..7 | 8..15
            v8bf up = *(const v8bf*)(base + 16 + hi * 8);   // K 16..23 | 24..31
            bfrag[nt] = __builtin_shufflevector(lo, up,
                0, 1, 2, 3, 4, 5, 6, 7, 8, 9, 10, 11, 12, 13, 14, 15);
        }

        // ---- 16 WMMAs per K-step ----
        #pragma unroll
        for (int mt = 0; mt < 4; ++mt)
            #pragma unroll
            for (int nt = 0; nt < 4; ++nt)
                acc[mt][nt] = __builtin_amdgcn_wmma_f32_16x16x32_bf16(
                    false, afrag[mt], false, bfrag[nt],
                    (short)0, acc[mt][nt], false, false);
        __syncthreads();
    }

    // ---- epilogue: C layout -> lanes 0-15 rows r, lanes 16-31 rows r+8 ----
    #pragma unroll
    for (int mt = 0; mt < 4; ++mt) {
        #pragma unroll
        for (int nt = 0; nt < 4; ++nt) {
            int n = n0 + wn * 64 + nt * 16 + lr;
            if (n < NO) {
                #pragma unroll
                for (int r = 0; r < 8; ++r) {
                    int m = m0 + wm * 64 + mt * 16 + r + hi * 8;
                    out[(size_t)m * NO + n] = acc[mt][nt][r];
                }
            }
        }
    }
}

// -------------------------------------------------------------------------
// Kernel 4: per-row softmax scaled by exp(log_library_size), in place.
// -------------------------------------------------------------------------
__global__ __launch_bounds__(256) void softmax_scale_kernel(
        float* __restrict__ out, const float* __restrict__ lls) {
    const int b   = blockIdx.x;
    const int tid = threadIdx.x;
    float* row = out + (size_t)b * NO;
    __shared__ float red[256];

    // pass 1: row max
    float m = -3.4e38f;
    for (int j = tid; j < NO; j += 256) m = fmaxf(m, row[j]);
    red[tid] = m;
    __syncthreads();
    for (int s = 128; s > 0; s >>= 1) {
        if (tid < s) red[tid] = fmaxf(red[tid], red[tid + s]);
        __syncthreads();
    }
    const float rowmax = red[0];
    __syncthreads();

    // pass 2: sum of exp
    float sum = 0.f;
    for (int j = tid; j < NO; j += 256) sum += __expf(row[j] - rowmax);
    red[tid] = sum;
    __syncthreads();
    for (int s = 128; s > 0; s >>= 1) {
        if (tid < s) red[tid] += red[tid + s];
        __syncthreads();
    }
    const float scale = __expf(lls[b]) / red[0];
    __syncthreads();

    // pass 3: write
    for (int j = tid; j < NO; j += 256) row[j] = __expf(row[j] - rowmax) * scale;
}

// -------------------------------------------------------------------------
extern "C" void kernel_launch(void* const* d_in, const int* in_sizes, int n_in,
                              void* d_out, int out_size, void* d_ws, size_t ws_size,
                              hipStream_t stream) {
    const float* z    = (const float*)d_in[0];   // (B, 2304)
    const float* lls  = (const float*)d_in[1];   // (B, 1)
    const float* cat  = (const float*)d_in[2];   // (B, 32)
    const float* mask = (const float*)d_in[3];   // (2048, 20000)
    const float* amask= (const float*)d_in[4];   // (256, 20000)
    const float* Wm   = (const float*)d_in[5];   // (2048, 20000)
    const float* Wa   = (const float*)d_in[6];   // (256, 20000)
    const float* Wc   = (const float*)d_in[7];   // (32, 20000)
    float* out = (float*)d_out;                  // (B, 20000)

    char* ws = (char*)d_ws;
    __bf16* Zb = (__bf16*)ws;                                    // 38,273,024 B
    __bf16* Wt = (__bf16*)(ws + (size_t)B_ROWS * KDIM * 2);      // 93,440,000 B

    // 1) pack zi -> bf16
    {
        size_t total = (size_t)B_ROWS * KDIM;
        int blocks = (int)((total + 255) / 256);
        pack_z_kernel<<<blocks, 256, 0, stream>>>(z, cat, Zb);
    }
    // 2) pack masked weights -> bf16, transposed to [NO x K]
    {
        dim3 grid(NO / 32, KDIM / 32);   // 625 x 73
        dim3 block(32, 8);
        pack_wt_kernel<<<grid, block, 0, stream>>>(mask, amask, Wm, Wa, Wc, Wt);
    }
    // 3) WMMA GEMM -> logits in d_out
    {
        dim3 grid((NO + TN - 1) / TN, B_ROWS / TM);  // 79 x 64
        gemm_kernel<<<grid, 256, 0, stream>>>(Zb, Wt, out);
    }
    // 4) softmax + library-size scaling, in place
    softmax_scale_kernel<<<B_ROWS, 256, 0, stream>>>(out, lls);
}
